// BertWhitenSVDBlock_85529978733272
// MI455X (gfx1250) — matrix operand order repaired and checked by reading, not compiled, not run on hardware
//
#include <hip/hip_runtime.h>
#include <hip/hip_bf16.h>
#include <math.h>

#if __has_include(<hip/amd_detail/amd_gfx1250_TDM.h>)
#define TDM_6ARG 1
#endif

typedef __attribute__((ext_vector_type(16))) _Float16 v16h;
typedef __attribute__((ext_vector_type(8)))  _Float16 v8h;
typedef __attribute__((ext_vector_type(8)))  float    v8f;
typedef __attribute__((ext_vector_type(4)))  unsigned int v4u;
typedef __attribute__((ext_vector_type(8)))  int v8i;
typedef __attribute__((ext_vector_type(4)))  int v4i;

#define WMMA_F16(a, b, c) \
  __builtin_amdgcn_wmma_f32_16x16x32_f16(false, (a), false, (b), (short)0, (c), false, false)

__device__ inline v8f vzero8() {
  v8f z = {0.f, 0.f, 0.f, 0.f, 0.f, 0.f, 0.f, 0.f};
  return z;
}
__device__ inline v8h ld8(const _Float16* p) { return *(const v8h*)p; }
__device__ inline void st8(_Float16* p, v8h v) { *(v8h*)p = v; }
__device__ inline v16h cat8(v8h lo, v8h hi) {
  return __builtin_shufflevector(lo, hi, 0, 1, 2, 3, 4, 5, 6, 7, 8, 9, 10, 11, 12, 13, 14, 15);
}

// A fragment (16x32): per lane two contiguous 8-half runs: [koff..koff+7], [koff+16..koff+23]
__device__ inline v16h load_a_frag_w(const _Float16* base, int lda, int row0, int lane) {
  int r = row0 + (lane & 15);
  int koff = (lane & 16) ? 8 : 0;
  const _Float16* p = base + (size_t)r * lda + koff;
  return cat8(ld8(p), ld8(p + 16));
}
// B fragment (32x16) from TRANSPOSED storage [col][k]: per lane two 8-half runs: [kst..kst+7], [kst+16..kst+23]
__device__ inline v16h load_bT_frag_w(const _Float16* baseT, int ldb, int col0, int lane) {
  int cc = col0 + (lane & 15);
  int kst = (lane & 16) ? 8 : 0;
  const _Float16* p = baseT + (size_t)cc * ldb + kst;
  return cat8(ld8(p), ld8(p + 16));
}

// ---------------------------------------------------------------------------
// TDM: async-load a 2D f16 tile (tile_w x tile_h, row stride = k_stride elems)
// from global into LDS with hardware padding of 4 DWORDs every 16 DWORDs
// (i.e. 32-half rows land on a 40-half pitch, matching As[64][40]).
// Issued per-wave (EXEC ignored); completion via TENSORcnt.
// ---------------------------------------------------------------------------
__device__ inline void tdm_load_tile_f16(unsigned lds_addr, const _Float16* gptr,
                                         unsigned k_stride, unsigned tile_w,
                                         unsigned tile_h) {
  unsigned long long ga = (unsigned long long)(size_t)gptr;
  v4u g0 = {1u,                                   // count=1, user descriptor
            lds_addr,                             // LDS byte address
            (unsigned)ga,                         // global_addr[31:0]
            (unsigned)(ga >> 32) | 0x80000000u};  // global_addr[56:32] | type=2
  unsigned td0 = k_stride;                        // tensor_dim0 (elems)
  unsigned td1 = 0x00FFFFFFu;                     // tensor_dim1 (large)
  unsigned long long st0 = k_stride;              // tensor_dim0_stride (elems)
  int d0 = (1 << 16)     // data_size = 2 bytes
         | (1 << 20)     // pad_enable
         | (3 << 22)     // pad_interval: 16 DWORDs
         | (3 << 25);    // pad_amount: 4 DWORDs
  v8i g1 = {d0,
            (int)((td0 & 0xFFFFu) << 16),                    // [47:32]=0, [63:48]=td0.lo
            (int)((td0 >> 16) | ((td1 & 0xFFFFu) << 16)),    // td0.hi | td1.lo
            (int)((td1 >> 16) | (tile_w << 16)),             // td1.hi | tile_dim0
            (int)tile_h,                                     // tile_dim1, tile_dim2=0
            (int)(st0 & 0xFFFFFFFFu),                        // stride0.lo
            (int)(st0 >> 32),                                // stride0.hi, stride1.lo=0
            0};
  v4i gz = {0, 0, 0, 0};
#if defined(TDM_6ARG)
  v8i gz8 = {0, 0, 0, 0, 0, 0, 0, 0};
  __builtin_amdgcn_tensor_load_to_lds(g0, g1, gz, gz, gz8, 0);
#else
  __builtin_amdgcn_tensor_load_to_lds(g0, g1, gz, gz, 0);
#endif
}

// ---------------------------------------------------------------------------
// Fused low-rank QKV: out = (x @ P_h) @ V_h + bias (Q scaled by 1/8).
// V output written TRANSPOSED as [B,H,DH,M] with wide stores.
// grid (64, 1, 48); block 128 (4 waves).
// ---------------------------------------------------------------------------
__global__ __launch_bounds__(128) void qkv_fused_kernel(
    const float* __restrict__ x,
    const float* __restrict__ Pq, const float* __restrict__ Vq, const float* __restrict__ bq,
    const float* __restrict__ Pk, const float* __restrict__ Vk, const float* __restrict__ bk,
    const float* __restrict__ Pv, const float* __restrict__ Vv, const float* __restrict__ bv,
    _Float16* __restrict__ Qb, _Float16* __restrict__ Kb, _Float16* __restrict__ VbT)
{
  constexpr int DM = 1024, RA = 32, DH = 64, MSEQ = 512, H = 16;
  __shared__ __align__(16) _Float16 As[64][40];    // x tile, row-major
  __shared__ __align__(16) _Float16 BsT[32][40];   // P tile, transposed [n][k]
  __shared__ __align__(16) _Float16 Qr[64][40];    // stage-1 result
  __shared__ __align__(16) _Float16 VhT[64][40];   // V_h transposed [d][r]

  const int tid = threadIdx.x, lane = tid & 31, w = tid >> 5;
  const int m0 = blockIdx.x * 64;
  const int z = blockIdx.z;
  const int which = z >> 4;  // 0=Q 1=K 2=V
  const int h = z & 15;
  const int c = lane & 15, hi2 = (lane >> 4) & 1;

  const float* Pm; const float* Vm; const float* bm; _Float16* Ob;
  if (which == 0)      { Pm = Pq; Vm = Vq; bm = bq; Ob = Qb;  }
  else if (which == 1) { Pm = Pk; Vm = Vk; bm = bk; Ob = Kb;  }
  else                 { Pm = Pv; Vm = Vv; bm = bv; Ob = VbT; }
  const float* Ph = Pm + (size_t)h * DM * RA;
  const float* Vh = Vm + (size_t)h * RA * DH;
  const float* bh = bm + h * DH;

  // stage V_h transposed into LDS (once)
  { int d = tid & 63, g0 = tid >> 6;
#pragma unroll
    for (int gg = g0; gg < 4; gg += 2) {
      v8h hb;
#pragma unroll
      for (int kk = 0; kk < 8; ++kk)
        hb[kk] = (_Float16)Vh[(size_t)(gg * 8 + kk) * DH + d];
      st8(&VhT[d][gg * 8], hb);
    }
  }

  v8f acc0 = vzero8(), acc1 = vzero8();

  // stage 1: q_r(64x32) = x_tile(64x1024) @ P_h(1024x32)
  for (int k0 = 0; k0 < DM; k0 += 32) {
    __syncthreads();
    { int r = tid >> 1, cb = (tid & 1) * 16;
      const float* src = x + (size_t)(m0 + r) * DM + k0 + cb;
      float4 f0 = *(const float4*)(src);
      float4 f1 = *(const float4*)(src + 4);
      float4 f2 = *(const float4*)(src + 8);
      float4 f3 = *(const float4*)(src + 12);
      v8h h0, h1;
      h0[0] = (_Float16)f0.x; h0[1] = (_Float16)f0.y; h0[2] = (_Float16)f0.z; h0[3] = (_Float16)f0.w;
      h0[4] = (_Float16)f1.x; h0[5] = (_Float16)f1.y; h0[6] = (_Float16)f1.z; h0[7] = (_Float16)f1.w;
      h1[0] = (_Float16)f2.x; h1[1] = (_Float16)f2.y; h1[2] = (_Float16)f2.z; h1[3] = (_Float16)f2.w;
      h1[4] = (_Float16)f3.x; h1[5] = (_Float16)f3.y; h1[6] = (_Float16)f3.z; h1[7] = (_Float16)f3.w;
      st8(&As[r][cb], h0);
      st8(&As[r][cb + 8], h1);
      if (k0 + 32 < DM) __builtin_prefetch(src + 32, 0, 3);
    }
    { int n = tid & 31, g = tid >> 5;  // transpose-stage P tile
      v8h hb;
#pragma unroll
      for (int kk = 0; kk < 8; ++kk)
        hb[kk] = (_Float16)Ph[(size_t)(k0 + g * 8 + kk) * RA + n];
      st8(&BsT[n][g * 8], hb);
    }
    __syncthreads();
    v16h a  = load_a_frag_w(&As[0][0], 40, w * 16, lane);
    v16h b0 = load_bT_frag_w(&BsT[0][0], 40, 0,  lane);
    v16h b1 = load_bT_frag_w(&BsT[0][0], 40, 16, lane);
    acc0 = WMMA_F16(a, b0, acc0);
    acc1 = WMMA_F16(a, b1, acc1);
  }
  __syncthreads();
  {
#pragma unroll
    for (int i = 0; i < 8; ++i) {
      int r = w * 16 + i + 8 * hi2;
      Qr[r][c]      = (_Float16)acc0[i];
      Qr[r][16 + c] = (_Float16)acc1[i];
    }
  }
  __syncthreads();

  // stage 2: out(16x64) = q_r(16x32) @ V_h(32x64) + bias
  {
    v16h aq = load_a_frag_w(&Qr[0][0], 40, w * 16, lane);
    const float qscale = (which == 0) ? 0.125f : 1.0f;
#pragma unroll
    for (int t = 0; t < 4; ++t) {
      v16h bf = load_bT_frag_w(&VhT[0][0], 40, t * 16, lane);
      v8f o = WMMA_F16(aq, bf, vzero8());
      float bval = bh[t * 16 + c];
      if (which == 2) {
        // transposed wide store: V^T[b][h][d][m], 8 consecutive tokens per lane
        v8h oh;
#pragma unroll
        for (int i = 0; i < 8; ++i) oh[i] = (_Float16)(o[i] + bval);
        int token0 = m0 + w * 16 + 8 * hi2;
        int bb = token0 >> 9, mm = token0 & 511;
        st8(Ob + (((size_t)bb * H + h) * DH + t * 16 + c) * MSEQ + mm, oh);
      } else {
#pragma unroll
        for (int i = 0; i < 8; ++i) {
          int token = m0 + w * 16 + i + 8 * hi2;
          int bb = token >> 9, mm = token & 511;
          float v = (o[i] + bval) * qscale;
          Ob[(((size_t)bb * H + h) * MSEQ + mm) * DH + t * 16 + c] = (_Float16)v;
        }
      }
    }
  }
}

// ---------------------------------------------------------------------------
// Flash attention, transposed formulation: S^T = K Q^T, O^T = V^T P^T.
// All WMMA operands are contiguous 128-bit loads (K,V^T direct from global/L2,
// P^T via per-wave LDS). Softmax stats are per-lane scalars (q = lane&15),
// combined with one shfl_xor(16). grid (8,16,8); block 128 (4 waves).
// ---------------------------------------------------------------------------
__global__ __launch_bounds__(128) void flash_attn_kernel(
    const _Float16* __restrict__ Qb, const _Float16* __restrict__ Kb,
    const _Float16* __restrict__ VbT, const float* __restrict__ mask,
    _Float16* __restrict__ ctx)
{
  constexpr int DH = 64, MSEQ = 512, H = 16;
  __shared__ __align__(16) _Float16 Ps[4][16][72];  // per-wave P^T: [q][kv]

  const int tid = threadIdx.x, lane = tid & 31, w = tid >> 5;
  const int b = blockIdx.z, h = blockIdx.y, m0 = blockIdx.x * 64;
  const int c = lane & 15, hi2 = (lane >> 4) & 1;
  const int kst16 = hi2 ? 8 : 0;  // B-frag k start
  const int koff8 = hi2 ? 8 : 0;   // A-frag k offset
  const size_t qkbase = ((size_t)b * H + h) * MSEQ;  // Q/K row base
  const size_t vtbase = ((size_t)b * H + h) * DH;    // V^T row base
  _Float16* psrow = &Ps[w][c][0];

  // Q as B operand (col = q), loaded once; k-dim = d (2 steps of 32)
  const _Float16* qp = Qb + (qkbase + m0 + w * 16 + c) * DH;
  v16h bq0 = cat8(ld8(qp + kst16), ld8(qp + kst16 + 16));
  v16h bq1 = cat8(ld8(qp + 32 + kst16), ld8(qp + 32 + kst16 + 16));

  float Mq = -1e30f, Lq = 0.f;
  v8f O[4];
#pragma unroll
  for (int t = 0; t < 4; ++t) O[t] = vzero8();

  for (int j = 0; j < MSEQ / 64; ++j) {
    const int n0 = j * 64;
    // S^T(64 kv x 16 q) = K(kv x d) @ Q^T(d x q)
    float st[4][8];
#pragma unroll
    for (int t = 0; t < 4; ++t) {
      const _Float16* kp = Kb + (qkbase + n0 + t * 16 + c) * DH;  // A row = kv
      v16h ak0 = cat8(ld8(kp + koff8), ld8(kp + koff8 + 16));
      v16h ak1 = cat8(ld8(kp + 32 + koff8), ld8(kp + 32 + koff8 + 16));
      v8f s = WMMA_F16(ak0, bq0, vzero8());
      s = WMMA_F16(ak1, bq1, s);
      const float* mp = mask + (size_t)b * MSEQ + n0 + t * 16 + 8 * hi2;
      float4 ml = *(const float4*)mp;
      float4 mh = *(const float4*)(mp + 4);
      st[t][0] = s[0] + ml.x; st[t][1] = s[1] + ml.y;
      st[t][2] = s[2] + ml.z; st[t][3] = s[3] + ml.w;
      st[t][4] = s[4] + mh.x; st[t][5] = s[5] + mh.y;
      st[t][6] = s[6] + mh.z; st[t][7] = s[7] + mh.w;
    }
    // online softmax: per-lane q = c; combine the two kv half-groups via xor-16
    float mloc = -1e30f;
#pragma unroll
    for (int t = 0; t < 4; ++t)
#pragma unroll
      for (int i = 0; i < 8; ++i) mloc = fmaxf(mloc, st[t][i]);
    mloc = fmaxf(mloc, __shfl_xor(mloc, 16, 32));
    float mnew = fmaxf(Mq, mloc);
    float sc = __expf(Mq - mnew);
    Mq = mnew;
    Lq *= sc;
#pragma unroll
    for (int t = 0; t < 4; ++t) O[t] *= sc;
    float rs = 0.f;
#pragma unroll
    for (int t = 0; t < 4; ++t) {
      v8h ph;
#pragma unroll
      for (int i = 0; i < 8; ++i) {
        float p = __expf(st[t][i] - mnew);
        rs += p;
        ph[i] = (_Float16)p;
      }
      st8(psrow + t * 16 + 8 * hi2, ph);  // P^T[q][kv], packed 128-bit store
    }
    rs += __shfl_xor(rs, 16, 32);
    Lq += rs;
    asm volatile("s_wait_dscnt 0x0" ::: "memory");  // wave-local LDS visibility

    // O^T(64 d x 16 q) += V^T(d x kv) @ P^T(kv x q)
    v16h bp0 = cat8(ld8(psrow + kst16), ld8(psrow + kst16 + 16));
    v16h bp1 = cat8(ld8(psrow + 32 + kst16), ld8(psrow + 32 + kst16 + 16));
#pragma unroll
    for (int t = 0; t < 4; ++t) {
      const _Float16* vp = VbT + (vtbase + t * 16 + c) * MSEQ + n0;  // A row = d
      v16h av0 = cat8(ld8(vp + koff8), ld8(vp + koff8 + 16));
      v16h av1 = cat8(ld8(vp + 32 + koff8), ld8(vp + 32 + koff8 + 16));
      O[t] = WMMA_F16(av0, bp0, O[t]);
      O[t] = WMMA_F16(av1, bp1, O[t]);
    }
  }

  // normalize (per-lane scalar 1/L) and write ctx with wide stores
  float inv = 1.0f / Lq;
  size_t tok = (size_t)b * MSEQ + m0 + w * 16 + c;
#pragma unroll
  for (int t = 0; t < 4; ++t) {
    v8h oh;
#pragma unroll
    for (int i = 0; i < 8; ++i) oh[i] = (_Float16)(O[t][i] * inv);
    st8((_Float16*)ctx + tok * (H * DH) + h * DH + t * 16 + 8 * hi2, oh);
  }
}

// ---------------------------------------------------------------------------
// Generic 64x64-tile WMMA GEMM: C = epi(A @ B + bias [+ res]).
// B staged transposed in LDS so all fragment loads are 128-bit.
// AF16 path: A tile staged by the Tensor Data Mover (async, TENSORcnt).
// ---------------------------------------------------------------------------
template <int AF16, int EPI, int OUTF16>
__global__ __launch_bounds__(128) void gemm_tile_kernel(
    const void* __restrict__ Ap, const float* __restrict__ Bp,
    const float* __restrict__ bias, const float* __restrict__ res,
    void* __restrict__ Cp, int M, int N, int K)
{
  __shared__ __align__(16) _Float16 As[64][40];
  __shared__ __align__(16) _Float16 BsT[64][40];
  const int tid = threadIdx.x, lane = tid & 31, w = tid >> 5;
  const int m0 = blockIdx.x * 64, n0 = blockIdx.y * 64;
  const int wr = (w >> 1) * 32, wc = (w & 1) * 32;
  const unsigned as_lds = (unsigned)(size_t)&As[0][0];  // LDS byte offset (low bits)

  v8f a00 = vzero8(), a01 = vzero8(), a10 = vzero8(), a11 = vzero8();

  for (int k0 = 0; k0 < K; k0 += 32) {
    __syncthreads();
    if (AF16) {
      // TDM: async DMA of the 64x32 f16 A tile into LDS (pads rows to 40 halves)
      if (tid < 32)
        tdm_load_tile_f16(as_lds, (const _Float16*)Ap + (size_t)m0 * K + k0,
                          (unsigned)K, 32u, 64u);
    } else {
      int r = tid >> 1, cb = (tid & 1) * 16;
      const float* A = (const float*)Ap + (size_t)(m0 + r) * K + k0 + cb;
      float4 f0 = *(const float4*)(A);
      float4 f1 = *(const float4*)(A + 4);
      float4 f2 = *(const float4*)(A + 8);
      float4 f3 = *(const float4*)(A + 12);
      v8h h0, h1;
      h0[0] = (_Float16)f0.x; h0[1] = (_Float16)f0.y; h0[2] = (_Float16)f0.z; h0[3] = (_Float16)f0.w;
      h0[4] = (_Float16)f1.x; h0[5] = (_Float16)f1.y; h0[6] = (_Float16)f1.z; h0[7] = (_Float16)f1.w;
      h1[0] = (_Float16)f2.x; h1[1] = (_Float16)f2.y; h1[2] = (_Float16)f2.z; h1[3] = (_Float16)f2.w;
      h1[4] = (_Float16)f3.x; h1[5] = (_Float16)f3.y; h1[6] = (_Float16)f3.z; h1[7] = (_Float16)f3.w;
      st8(&As[r][cb], h0);
      st8(&As[r][cb + 8], h1);
      if (k0 + 32 < K) __builtin_prefetch(A + 32, 0, 3);
    }
    { int n = tid & 63, g0 = tid >> 6;  // transpose-stage B (coalesced per-k row reads)
#pragma unroll
      for (int gg = g0; gg < 4; gg += 2) {
        v8h hb;
#pragma unroll
        for (int kk = 0; kk < 8; ++kk)
          hb[kk] = (_Float16)Bp[(size_t)(k0 + gg * 8 + kk) * N + n0 + n];
        st8(&BsT[n][gg * 8], hb);
      }
      if (k0 + 32 < K) __builtin_prefetch(Bp + (size_t)(k0 + 32 + g0) * N + n0 + n, 0, 3);
    }
    if (AF16) {
      if (tid < 32) __builtin_amdgcn_s_wait_tensorcnt(0);  // wave 0 owns TENSORcnt
    }
    __syncthreads();
    v16h fa0 = load_a_frag_w(&As[0][0], 40, wr,      lane);
    v16h fa1 = load_a_frag_w(&As[0][0], 40, wr + 16, lane);
    v16h fb0 = load_bT_frag_w(&BsT[0][0], 40, wc,      lane);
    v16h fb1 = load_bT_frag_w(&BsT[0][0], 40, wc + 16, lane);
    a00 = WMMA_F16(fa0, fb0, a00);
    a01 = WMMA_F16(fa0, fb1, a01);
    a10 = WMMA_F16(fa1, fb0, a10);
    a11 = WMMA_F16(fa1, fb1, a11);
  }

  const int c = lane & 15, hi2 = (lane >> 4) & 1;
  v8f accs[4] = {a00, a01, a10, a11};
#pragma unroll
  for (int fr = 0; fr < 2; ++fr) {
#pragma unroll
    for (int fc = 0; fc < 2; ++fc) {
      v8f acc = accs[fr * 2 + fc];
#pragma unroll
      for (int i = 0; i < 8; ++i) {
        int row = m0 + wr + fr * 16 + i + 8 * hi2;
        int col = n0 + wc + fc * 16 + c;
        float v = acc[i];
        if (EPI != 0) v += bias[col];
        if (EPI == 3) v += res[(size_t)row * N + col];
        if (EPI == 2) v = 0.5f * v * (1.0f + erff(v * 0.70710678118f));  // exact GELU
        if (OUTF16) ((_Float16*)Cp)[(size_t)row * N + col] = (_Float16)v;
        else        ((float*)Cp)[(size_t)row * N + col] = v;
      }
    }
  }
}

// ---------------------------------------------------------------------------
// LayerNorm over D=1024, one 256-thread block per token row, float4 I/O.
// ---------------------------------------------------------------------------
__global__ __launch_bounds__(256) void layernorm_kernel(
    const float* __restrict__ in, const float* __restrict__ g,
    const float* __restrict__ bta, float* __restrict__ out)
{
  constexpr int D = 1024;
  __shared__ float red[256];
  const int tid = threadIdx.x;
  const float4 u = ((const float4*)(in + (size_t)blockIdx.x * D))[tid];
  red[tid] = u.x + u.y + u.z + u.w;
  __syncthreads();
  for (int o = 128; o > 0; o >>= 1) { if (tid < o) red[tid] += red[tid + o]; __syncthreads(); }
  float mu = red[0] * (1.0f / D);
  __syncthreads();
  float d0 = u.x - mu, d1 = u.y - mu, d2 = u.z - mu, d3 = u.w - mu;
  red[tid] = d0 * d0 + d1 * d1 + d2 * d2 + d3 * d3;
  __syncthreads();
  for (int o = 128; o > 0; o >>= 1) { if (tid < o) red[tid] += red[tid + o]; __syncthreads(); }
  float rstd = rsqrtf(red[0] * (1.0f / D) + 1e-12f);
  const float4 gv = ((const float4*)g)[tid];
  const float4 bv = ((const float4*)bta)[tid];
  float4 r;
  r.x = d0 * rstd * gv.x + bv.x;
  r.y = d1 * rstd * gv.y + bv.y;
  r.z = d2 * rstd * gv.z + bv.z;
  r.w = d3 * rstd * gv.w + bv.w;
  ((float4*)(out + (size_t)blockIdx.x * D))[tid] = r;
}

// ---------------------------------------------------------------------------
extern "C" void kernel_launch(void* const* d_in, const int* in_sizes, int n_in,
                              void* d_out, int out_size, void* d_ws, size_t ws_size,
                              hipStream_t stream)
{
  (void)in_sizes; (void)n_in; (void)out_size; (void)ws_size;
  const float* x    = (const float*)d_in[0];
  const float* mask = (const float*)d_in[1];
  const float* Pq   = (const float*)d_in[2];
  const float* Vq   = (const float*)d_in[3];
  const float* Pk   = (const float*)d_in[4];
  const float* Vk   = (const float*)d_in[5];
  const float* Pv   = (const float*)d_in[6];
  const float* Vv   = (const float*)d_in[7];
  const float* bq   = (const float*)d_in[8];
  const float* bk   = (const float*)d_in[9];
  const float* bv   = (const float*)d_in[10];
  const float* Uo   = (const float*)d_in[11];
  const float* Vo   = (const float*)d_in[12];
  const float* bo   = (const float*)d_in[13];
  const float* U1   = (const float*)d_in[14];
  const float* V1   = (const float*)d_in[15];
  const float* b1   = (const float*)d_in[16];
  const float* U2   = (const float*)d_in[17];
  const float* V2   = (const float*)d_in[18];
  const float* b2   = (const float*)d_in[19];
  const float* g1   = (const float*)d_in[20];
  const float* be1  = (const float*)d_in[21];
  const float* g2   = (const float*)d_in[22];
  const float* be2  = (const float*)d_in[23];

  char* ws = (char*)d_ws;
  const size_t T = 4096;
  size_t off = 0;
  auto take = [&](size_t bytes) -> char* {
    char* p = ws + off;
    off += (bytes + 255) & ~(size_t)255;
    return p;
  };
  _Float16* Qb  = (_Float16*)take(T * 1024 * sizeof(_Float16));  // [B,H,M,DH]
  _Float16* Kb  = (_Float16*)take(T * 1024 * sizeof(_Float16));  // [B,H,M,DH]
  _Float16* VbT = (_Float16*)take(T * 1024 * sizeof(_Float16));  // [B,H,DH,M] (transposed)
  _Float16* ctx = (_Float16*)take(T * 1024 * sizeof(_Float16));  // [T, DM]
  float*    tmp = (float*)take(T * 384 * sizeof(float));         // rank buffer, reused 3x
  float*    pre = (float*)take(T * 1024 * sizeof(float));        // pre-LN buffer, reused
  float*    x1  = (float*)take(T * 1024 * sizeof(float));
  _Float16* hdn = (_Float16*)take(T * 4096 * sizeof(_Float16));

  // 1) fused low-rank QKV (Q scaled by 1/sqrt(dh); V stored transposed)
  qkv_fused_kernel<<<dim3(64, 1, 48), 128, 0, stream>>>(
      x, Pq, Vq, bq, Pk, Vk, bk, Pv, Vv, bv, Qb, Kb, VbT);
  // 2) flash attention -> ctx (f16)
  flash_attn_kernel<<<dim3(8, 16, 8), 128, 0, stream>>>(Qb, Kb, VbT, mask, ctx);
  // 3) ctx @ Uo -> tmp [T,384]   (A tile via TDM)
  gemm_tile_kernel<1, 0, 0><<<dim3(64, 6), 128, 0, stream>>>(
      ctx, Uo, nullptr, nullptr, tmp, 4096, 384, 1024);
  // 4) tmp @ Vo + bo + x -> pre
  gemm_tile_kernel<0, 3, 0><<<dim3(64, 16), 128, 0, stream>>>(
      tmp, Vo, bo, x, pre, 4096, 1024, 384);
  // 5) LN1 -> x1
  layernorm_kernel<<<dim3(4096), 256, 0, stream>>>(pre, g1, be1, x1);
  // 6) x1 @ U1 -> tmp
  gemm_tile_kernel<0, 0, 0><<<dim3(64, 6), 128, 0, stream>>>(
      x1, U1, nullptr, nullptr, tmp, 4096, 384, 1024);
  // 7) gelu(tmp @ V1 + b1) -> hdn (f16)
  gemm_tile_kernel<0, 2, 1><<<dim3(64, 64), 128, 0, stream>>>(
      tmp, V1, b1, nullptr, hdn, 4096, 4096, 384);
  // 8) hdn @ U2 -> tmp   (A tile via TDM)
  gemm_tile_kernel<1, 0, 0><<<dim3(64, 6), 128, 0, stream>>>(
      hdn, U2, nullptr, nullptr, tmp, 4096, 384, 4096);
  // 9) tmp @ V2 + b2 + x1 -> pre
  gemm_tile_kernel<0, 3, 0><<<dim3(64, 16), 128, 0, stream>>>(
      tmp, V2, b2, x1, pre, 4096, 1024, 384);
  // 10) LN2 -> out
  layernorm_kernel<<<dim3(4096), 256, 0, stream>>>(pre, g2, be2, (float*)d_out);
}
